// MsaHmmLayer_919123002048
// MI455X (gfx1250) — matrix operand order repaired and checked by reading, not compiled, not run on hardware
//
#include <hip/hip_runtime.h>
#include <math.h>
#include <stdint.h>

#define MDL   4
#define BATCH 32
#define TLEN  512
#define NSYM  26
#define NQ    512

typedef _Float16 v16h __attribute__((ext_vector_type(16)));
typedef float    v8f  __attribute__((ext_vector_type(8)));

// p (rescaled alpha, f16) is stored in LDS directly in WMMA A-fragment order.
// Assumed 16-bit A-matrix 16x32 layout (ISA 7.12.2): lane = (row%16) + 16*khi,
// lanes<16 hold klocal {0-7,16-23}, lanes>=16 hold {8-15,24-31}.
__device__ __forceinline__ int pidx(int b, int i) {
  int rt   = b >> 4;            // row tile (batch 0-15 / 16-31)
  int lr   = b & 15;
  int ks   = i >> 5;            // K-step of 32
  int kl   = i & 31;
  int khi  = (kl >> 3) & 1;
  int h    = (kl & 7) + ((kl >> 4) << 3);
  int lane = lr + (khi << 4);
  return (((ks * 2 + rt) * 32 + lane) * 16) + h;
}

// ---------------------------------------------------------------------------
// Prep 1: A = softmax(trans_logits) per row, written as f16 WMMA B-fragments.
// B-matrix 32x16 assumed layout: element (klocal, n): lane = n + 16*((klocal/8)%2),
// half = klocal%8 + 8*(klocal/16). Fragment index: (((m*32+ct)*16+ks)*32+lane)*16+h.
// ---------------------------------------------------------------------------
__global__ void prep_trans(const float* __restrict__ tl, _Float16* __restrict__ Afrag) {
  __shared__ float red[256];
  int row = blockIdx.x;                // m*NQ + i
  int m = row >> 9, i = row & 511;
  const float* src = tl + (size_t)row * NQ;
  int t = threadIdx.x;
  float x0 = src[t], x1 = src[t + 256];
  red[t] = fmaxf(x0, x1);
  __syncthreads();
  for (int s = 128; s > 0; s >>= 1) {
    if (t < s) red[t] = fmaxf(red[t], red[t + s]);
    __syncthreads();
  }
  float mx = red[0];
  __syncthreads();
  float e0 = __expf(x0 - mx), e1 = __expf(x1 - mx);
  red[t] = e0 + e1;
  __syncthreads();
  for (int s = 128; s > 0; s >>= 1) {
    if (t < s) red[t] += red[t + s];
    __syncthreads();
  }
  float rs = 1.0f / red[0];
  int ks = i >> 5, kl = i & 31;
  for (int rep = 0; rep < 2; ++rep) {
    int j = t + rep * 256;
    float val = (rep ? e1 : e0) * rs;
    int ct = j >> 4, n = j & 15;
    int lane = n + (((kl >> 3) & 1) << 4);
    int h = (kl & 7) + ((kl >> 4) << 3);
    Afrag[((((size_t)m * 32 + ct) * 16 + ks) * 32 + lane) * 16 + h] = (_Float16)val;
  }
}

// ---------------------------------------------------------------------------
// Prep 2: Bprob = softmax(emission_logits) over S=26, f32, layout [m][q][s].
// ---------------------------------------------------------------------------
__global__ void prep_emit(const float* __restrict__ el, float* __restrict__ Bp) {
  int row  = blockIdx.x * 4 + (threadIdx.x >> 5);   // m*NQ + q
  int lane = threadIdx.x & 31;
  const float* src = el + (size_t)row * NSYM;
  float x = (lane < NSYM) ? src[lane] : -1e30f;
  float mx = x;
  for (int s = 16; s; s >>= 1) mx = fmaxf(mx, __shfl_xor(mx, s, 32));
  float e = (lane < NSYM) ? __expf(x - mx) : 0.0f;
  float sm = e;
  for (int s = 16; s; s >>= 1) sm += __shfl_xor(sm, s, 32);
  if (lane < NSYM) Bp[(size_t)row * NSYM + lane] = e / sm;
}

// ---------------------------------------------------------------------------
// Prep 3: logpi = log_softmax(init_logits) per model.
// ---------------------------------------------------------------------------
__global__ void prep_init(const float* __restrict__ il, float* __restrict__ logpi) {
  __shared__ float red[512];
  int m = blockIdx.x, t = threadIdx.x;
  float x = il[m * NQ + t];
  red[t] = x;
  __syncthreads();
  for (int s = 256; s > 0; s >>= 1) {
    if (t < s) red[t] = fmaxf(red[t], red[t + s]);
    __syncthreads();
  }
  float mx = red[0];
  __syncthreads();
  red[t] = __expf(x - mx);
  __syncthreads();
  for (int s = 256; s > 0; s >>= 1) {
    if (t < s) red[t] += red[t + s];
    __syncthreads();
  }
  logpi[m * NQ + t] = x - (mx + __logf(red[0]));
}

// ---------------------------------------------------------------------------
// Main kernel: one 512-thread workgroup (16 waves) per model.
// Wave w owns 32 output columns (col-tiles 2w, 2w+1): 4 WMMA accumulators,
// each LDS p-fragment feeds two WMMAs. Matrix B-fragments stream from L2
// each step as global_load_b128 (opaque per-iteration index offset stops
// LICM register-caching; pointer provenance kept so loads stay GLOBAL).
// ---------------------------------------------------------------------------
__global__ __launch_bounds__(512)
void hmm_fwd(const _Float16* __restrict__ Afrag, const float* __restrict__ Bp,
             const float* __restrict__ logpi, const int* __restrict__ seqs,
             float* __restrict__ out) {
  __shared__ __attribute__((aligned(32))) _Float16 pfrag[16 * 2 * 32 * 16]; // 32 KB
  __shared__ float        BpL[NSYM][520];   // emission probs, [sym][state]
  __shared__ unsigned int mx2u[2][BATCH];   // per-row max (bit pattern, >=0)
  __shared__ float        Lrow[BATCH];      // running log-scale per row
  __shared__ float        recip[BATCH];     // 1/max per row
  __shared__ int          symb[BATCH];      // current step's symbols

  const int m     = blockIdx.x;
  const int tid   = threadIdx.x;
  const int lane  = tid & 31;
  const int wave  = tid >> 5;               // 0..15
  const int ct0   = wave * 2;               // two column tiles per wave
  const int ct1   = wave * 2 + 1;
  const int nhalf = lane >> 4;
  const int ncol  = lane & 15;
  const int jcol0 = ct0 * 16 + ncol;
  const int jcol1 = ct1 * 16 + ncol;

  const _Float16* gAm = Afrag + (size_t)m * NQ * NQ;
  const v16h* pfv = (const v16h*)pfrag;

  // ---- stage Bprob into LDS (transposed [s][q]) + init small arrays ----
  for (int idx = tid; idx < NQ * NSYM; idx += 512) {
    int q = idx / NSYM, s = idx - q * NSYM;
    BpL[s][q] = Bp[((size_t)m * NQ + q) * NSYM + s];
  }
  if (tid < BATCH) {
    mx2u[0][tid] = 0u;
    mx2u[1][tid] = 0u;
    symb[tid] = seqs[((size_t)m * BATCH + tid) * TLEN + 0];
  }
  __syncthreads();

  // ---- t = 0: alpha0 = logpi[j] + log Bprob[j][s_b]; wave handles 2 batches ----
  for (int half = 0; half < 2; ++half) {
    int b = wave + 16 * half;
    int s0 = symb[b];
    float vals[16];
    float mx = -1e30f;
    for (int r = 0; r < 16; ++r) {
      int j = lane + 32 * r;
      float a = logpi[m * NQ + j] + __logf(BpL[s0][j]);
      vals[r] = a;
      mx = fmaxf(mx, a);
    }
    for (int s = 16; s; s >>= 1) mx = fmaxf(mx, __shfl_xor(mx, s, 32));
    for (int r = 0; r < 16; ++r) {
      int j = lane + 32 * r;
      pfrag[pidx(b, j)] = (_Float16)__expf(vals[r] - mx);
    }
    if (lane == 0) Lrow[b] = mx;
  }
  __syncthreads();
  if (tid < BATCH) symb[tid] = seqs[((size_t)m * BATCH + tid) * TLEN + 1];
  __syncthreads();

  // ---- forward recursion ----
  for (int t = 1; t < TLEN; ++t) {
    const int cur = t & 1;

    // Opaque per-iteration zero: index is not loop-invariant (no LICM of the
    // loads -> no 256-VGPR caching/spill), but the base pointer keeps its
    // global address space, so these stay global_load_b128 from L2.
    int zero = 0;
    asm volatile("" : "+s"(zero));
    const v16h* g0 = (const v16h*)gAm + (size_t)(ct0 * 512 + zero);
    const v16h* g1 = (const v16h*)gAm + (size_t)(ct1 * 512 + zero);

    // phase 1: q = p * A for this wave's 32 columns (4 tiles, K = 512)
    v8f acc00 = {}, acc10 = {}, acc01 = {}, acc11 = {};
#pragma unroll
    for (int ks = 0; ks < 16; ++ks) {
      v16h a0 = pfv[(ks * 2 + 0) * 32 + lane];
      v16h a1 = pfv[(ks * 2 + 1) * 32 + lane];
      v16h b0 = g0[ks * 32 + lane];
      v16h b1 = g1[ks * 32 + lane];
      acc00 = __builtin_amdgcn_wmma_f32_16x16x32_f16(false, a0, false, b0,
                                                     (short)0, acc00, false, false);
      acc10 = __builtin_amdgcn_wmma_f32_16x16x32_f16(false, a1, false, b0,
                                                     (short)0, acc10, false, false);
      acc01 = __builtin_amdgcn_wmma_f32_16x16x32_f16(false, a0, false, b1,
                                                     (short)0, acc01, false, false);
      acc11 = __builtin_amdgcn_wmma_f32_16x16x32_f16(false, a1, false, b1,
                                                     (short)0, acc11, false, false);
    }

    // phase 2: multiply by emission probability (elementwise on D tiles)
    float v00[8], v10[8], v01[8], v11[8];
#pragma unroll
    for (int v = 0; v < 8; ++v) {
      int r0 = v + 8 * nhalf;           // D layout: lanes<16 rows v, lanes>=16 rows v+8
      int r1 = r0 + 16;
      int s0 = symb[r0], s1 = symb[r1];
      v00[v] = acc00[v] * BpL[s0][jcol0];
      v01[v] = acc01[v] * BpL[s0][jcol1];
      v10[v] = acc10[v] * BpL[s1][jcol0];
      v11[v] = acc11[v] * BpL[s1][jcol1];
    }

    // phase 3: per-row max across both col-tiles + 16 lanes, then LDS atomic max
#pragma unroll
    for (int v = 0; v < 8; ++v) {
      float m0 = fmaxf(v00[v], v01[v]);
      float m1 = fmaxf(v10[v], v11[v]);
      for (int s = 8; s; s >>= 1) {
        m0 = fmaxf(m0, __shfl_xor(m0, s, 32));
        m1 = fmaxf(m1, __shfl_xor(m1, s, 32));
      }
      if (ncol == 0) {   // lanes 0 and 16
        int r0 = v + 8 * nhalf;
        atomicMax(&mx2u[cur][r0],      __float_as_uint(m0));
        atomicMax(&mx2u[cur][r0 + 16], __float_as_uint(m1));
      }
    }
    __syncthreads();

    // phase 5: update log-scale, reciprocal, reset other buffer, fetch next syms
    if (tid < BATCH) {
      float mxv = fmaxf(__uint_as_float(mx2u[cur][tid]), 1e-30f);
      Lrow[tid] += __logf(mxv);
      recip[tid] = 1.0f / mxv;
      mx2u[cur ^ 1][tid] = 0u;
      if (t + 1 < TLEN)
        symb[tid] = seqs[((size_t)m * BATCH + tid) * TLEN + (t + 1)];
    }
    __syncthreads();

    // phase 7: p_new = (q * Bprob) / rowmax -> f16, scattered into frag layout
#pragma unroll
    for (int v = 0; v < 8; ++v) {
      int r0 = v + 8 * nhalf;
      int r1 = r0 + 16;
      float rc0 = recip[r0], rc1 = recip[r1];
      pfrag[pidx(r0, jcol0)] = (_Float16)(v00[v] * rc0);
      pfrag[pidx(r0, jcol1)] = (_Float16)(v01[v] * rc0);
      pfrag[pidx(r1, jcol0)] = (_Float16)(v10[v] * rc1);
      pfrag[pidx(r1, jcol1)] = (_Float16)(v11[v] * rc1);
    }
    __syncthreads();
  }

  // ---- finish: loglik_b = L_b + log(sum_j p_j); wave reduces 2 batches ----
  for (int half = 0; half < 2; ++half) {
    int b = wave + 16 * half;
    float sum = 0.0f;
    for (int r = 0; r < 16; ++r) {
      int j = lane + 32 * r;
      sum += (float)pfrag[pidx(b, j)];
    }
    for (int s = 16; s; s >>= 1) sum += __shfl_xor(sum, s, 32);
    if (lane == 0) out[m * BATCH + b] = Lrow[b] + __logf(sum);
  }
}

// ---------------------------------------------------------------------------
extern "C" void kernel_launch(void* const* d_in, const int* in_sizes, int n_in,
                              void* d_out, int out_size, void* d_ws, size_t ws_size,
                              hipStream_t stream) {
  (void)in_sizes; (void)n_in; (void)out_size; (void)ws_size;
  const float* trans = (const float*)d_in[0];
  const float* emis  = (const float*)d_in[1];
  const float* initl = (const float*)d_in[2];
  const int*   seqs  = (const int*)d_in[3];
  float* out = (float*)d_out;

  char* ws = (char*)d_ws;
  _Float16* Afrag = (_Float16*)ws;                                   // 2 MB
  float*    Bp    = (float*)(ws + (size_t)MDL * NQ * NQ * 2);        // 208 KB
  float*    lpi   = (float*)(ws + (size_t)MDL * NQ * NQ * 2
                                + (size_t)MDL * NQ * NSYM * 4);      // 8 KB

  prep_trans<<<MDL * NQ, 256, 0, stream>>>(trans, Afrag);
  prep_emit<<<(MDL * NQ) / 4, 128, 0, stream>>>(emis, Bp);
  prep_init<<<MDL, NQ, 0, stream>>>(initl, lpi);
  hmm_fwd<<<MDL, 512, 0, stream>>>(Afrag, Bp, lpi, seqs, out);
}
